// Attention_60584808677905
// MI455X (gfx1250) — matrix operand order repaired and checked by reading, not compiled, not run on hardware
//
#include <hip/hip_runtime.h>
#include <hip/hip_bf16.h>

typedef __bf16 bf16;
typedef __bf16 v16bf __attribute__((ext_vector_type(16)));
typedef __bf16 v8bf  __attribute__((ext_vector_type(8)));
typedef __bf16 v4bf  __attribute__((ext_vector_type(4)));
typedef __bf16 v2bf  __attribute__((ext_vector_type(2)));
typedef float  v8f   __attribute__((ext_vector_type(8)));

union Frag16 { v16bf v; v8bf h[2]; };

static constexpr int Bc  = 2;
static constexpr int Sc  = 2048;
static constexpr int Dm  = 4096;
static constexpr int NH  = 32;
static constexpr int NKV = 8;
static constexpr int HD  = 128;
static constexpr int NREP = NH / NKV;
static constexpr int BS  = Bc * Sc;          // 4096 rows

// ---------------------------------------------------------------------------
// CDNA5 async global->LDS copy (ASYNCcnt-tracked DMA, no VGPR staging).
// GV mode: vdst = LDS byte address, vaddr = 64-bit global address.
// LDS aperture: flat_addr[31:0] is the LDS byte address.
// ---------------------------------------------------------------------------
__device__ __forceinline__ void async_copy_b128(const bf16* g, bf16* l) {
  unsigned lds = (unsigned)(size_t)l;
  asm volatile("global_load_async_to_lds_b128 %0, %1, off"
               :: "v"(lds), "v"(g) : "memory");
}
__device__ __forceinline__ void wait_async() {
  asm volatile("s_wait_asynccnt 0x0" ::: "memory");
}

// ---------------------------------------------------------------------------
// Elementwise f32 -> bf16 (vectorized float4 -> 4x bf16).
// ---------------------------------------------------------------------------
__global__ __launch_bounds__(256) void cvt_bf16(
    const float* __restrict__ in, bf16* __restrict__ out)
{
  size_t i = ((size_t)blockIdx.x * 256 + threadIdx.x) * 4;
  const float4 f = *(const float4*)(in + i);
  v4bf w; w.x = (bf16)f.x; w.y = (bf16)f.y; w.z = (bf16)f.z; w.w = (bf16)f.w;
  *(v4bf*)(out + i) = w;
}

// ---------------------------------------------------------------------------
// Weight convert + transpose: W f32 [K][N] -> WT bf16 [N][K].
// LDS-tiled 32x32 so both global read and write are coalesced.
// ---------------------------------------------------------------------------
__global__ __launch_bounds__(256) void transpose_cvt_w(
    const float* __restrict__ W, bf16* __restrict__ WT, int K, int N)
{
  __shared__ bf16 tile[32][33];
  const int bn = blockIdx.x * 32;           // n base
  const int bk = blockIdx.y * 32;           // k base
  const int tx = threadIdx.x & 31;
  const int ty = threadIdx.x >> 5;          // 0..7
#pragma unroll
  for (int i = 0; i < 32; i += 8)
    tile[ty + i][tx] = (bf16)W[(size_t)(bk + ty + i) * N + bn + tx];
  __syncthreads();
#pragma unroll
  for (int i = 0; i < 32; i += 8)
    WT[(size_t)(bn + ty + i) * K + bk + tx] = tile[tx][ty + i];
}

// ---------------------------------------------------------------------------
// GEMM: C[M,N] = A[M,K] @ B[K,N] with B pre-transposed (Bt = [N][K]), bf16 in,
// f32 out. 256 threads = 8 waves; tile 128x128, K-step 32 (one WMMA depth).
// Double-buffered LDS fed by global_load_async_to_lds_b128; all fragments
// preloaded so the 8 WMMAs issue as one burst.
// ---------------------------------------------------------------------------
#define GBM 128
#define GBN 128
#define GBK 32

__global__ __launch_bounds__(256) void gemm_bf16(
    const bf16* __restrict__ A, const bf16* __restrict__ Bt,
    float* __restrict__ C, int M, int N, int K)
{
  __shared__ __align__(32) bf16 As[2][GBM][GBK];   // [m][k]
  __shared__ __align__(32) bf16 Bs[2][GBN][GBK];   // [n][k]

  const int tid  = threadIdx.x;
  const int wave = tid >> 5;
  const int lane = tid & 31;
  const int hl   = lane >> 4;
  const int l16  = lane & 15;
  const int wm   = (wave >> 1) * 32;
  const int wn   = (wave & 1) * 64;
  const int rowBase = blockIdx.y * GBM;
  const int colBase = blockIdx.x * GBN;

  v8f acc[2][4];
#pragma unroll
  for (int i = 0; i < 2; ++i)
#pragma unroll
    for (int j = 0; j < 4; ++j) acc[i][j] = (v8f)0.0f;

  // prologue: async-issue tile 0 (each tile = 512 x 16B chunks, 2 per thread)
#pragma unroll
  for (int i = 0; i < 2; ++i) {
    int c = tid + i * 256;
    int r = c >> 2, cc = (c & 3) << 3;
    async_copy_b128(A  + (size_t)(rowBase + r) * K + cc, &As[0][r][cc]);
    async_copy_b128(Bt + (size_t)(colBase + r) * K + cc, &Bs[0][r][cc]);
  }

  int buf = 0;
  for (int k0 = 0; k0 < K; k0 += GBK) {
    wait_async();          // my DMA into As[buf]/Bs[buf] landed
    __syncthreads();       // everyone's landed; everyone done reading buf^1

    if (k0 + GBK < K) {    // overlap next tile's DMA with this tile's WMMAs
      int nb = buf ^ 1;
#pragma unroll
      for (int i = 0; i < 2; ++i) {
        int c = tid + i * 256;
        int r = c >> 2, cc = (c & 3) << 3;
        async_copy_b128(A  + (size_t)(rowBase + r) * K + k0 + GBK + cc, &As[nb][r][cc]);
        async_copy_b128(Bt + (size_t)(colBase + r) * K + k0 + GBK + cc, &Bs[nb][r][cc]);
      }
    }

    // preload ALL fragments, then issue the 8 WMMAs back-to-back
    Frag16 a[2], bfr[4];
#pragma unroll
    for (int mt = 0; mt < 2; ++mt) {
      int m = wm + mt * 16 + l16;
      a[mt].v = *(const v16bf*)&As[buf][m][hl * 16];         // K = hl*16 .. +16
    }
#pragma unroll
    for (int nt = 0; nt < 4; ++nt) {
      int n = wn + nt * 16 + l16;
      bfr[nt].v = *(const v16bf*)&Bs[buf][n][hl * 16];       // K = hl*16 .. +16
    }
#pragma unroll
    for (int nt = 0; nt < 4; ++nt)
#pragma unroll
      for (int mt = 0; mt < 2; ++mt) {
        acc[mt][nt] = __builtin_amdgcn_wmma_f32_16x16x32_bf16(
            false, a[mt].v, false, bfr[nt].v, (short)0, acc[mt][nt], false, false);
      }
    buf ^= 1;
  }

#pragma unroll
  for (int mt = 0; mt < 2; ++mt)
#pragma unroll
    for (int nt = 0; nt < 4; ++nt)
#pragma unroll
      for (int r = 0; r < 8; ++r) {
        int row = rowBase + wm + mt * 16 + hl * 8 + r;
        int col = colBase + wn + nt * 16 + l16;
        C[(size_t)row * N + col] = acc[mt][nt][r];
      }
}

// ---------------------------------------------------------------------------
// RoPE + head transpose:  in f32 [B][S][H][HD]  ->  out bf16 [B][H][S][HD]
// ---------------------------------------------------------------------------
__global__ __launch_bounds__(256) void rope_transpose(
    const float* __restrict__ in, const float* __restrict__ cf,
    const float* __restrict__ sf, bf16* __restrict__ out, int H, int applyRope)
{
  size_t idx = (size_t)blockIdx.x * 256 + threadIdx.x;
  int i = (int)(idx & 63);  size_t t = idx >> 6;
  int h = (int)(t % H);     t /= H;
  int s = (int)(t % Sc);    int b = (int)(t / Sc);

  const float2 f = *(const float2*)(in + (((size_t)(b * Sc + s) * H + h) * HD) + 2 * i);
  float orr = f.x, oii = f.y;
  if (applyRope) {
    float c = cf[s * 64 + i], sn = sf[s * 64 + i];
    orr = f.x * c - f.y * sn;
    oii = f.x * sn + f.y * c;
  }
  v2bf w; w.x = (bf16)orr; w.y = (bf16)oii;
  *(v2bf*)(out + (((size_t)(b * H + h) * Sc + s) * HD) + 2 * i) = w;
}

// V: in f32 [B][S][NKV][HD] -> out bf16 [B][NKV][HD][S]  (d-major for B-frags)
__global__ __launch_bounds__(256) void v_transpose(
    const float* __restrict__ in, bf16* __restrict__ out)
{
  size_t idx = (size_t)blockIdx.x * 256 + threadIdx.x;
  int d = (int)(idx & (HD - 1));  size_t t = idx >> 7;
  int g = (int)(t % NKV);         t /= NKV;
  int s = (int)(t % Sc);          int b = (int)(t / Sc);
  float v = in[(((size_t)(b * Sc + s) * NKV + g) * HD) + d];
  out[(((size_t)(b * NKV + g) * HD + d) * Sc) + s] = (bf16)v;
}

// ---------------------------------------------------------------------------
// Flash attention, causal, online softmax. 128 threads = 4 waves.
// Tile 64 q-rows x 64 kv-cols; K/V tiles double-buffered via async LDS DMA
// so the next tile's DMA overlaps this tile's 32 WMMAs + softmax.
// ---------------------------------------------------------------------------
__global__ __launch_bounds__(128) void flash_attn(
    const bf16* __restrict__ qT,   // [B][NH][S][HD]
    const bf16* __restrict__ kT,   // [B][NKV][S][HD]
    const bf16* __restrict__ vT,   // [B][NKV][HD][S]
    bf16* __restrict__ ctx)        // [B][S][NH*HD]
{
  __shared__ __align__(32) bf16 Qs[64][HD];
  __shared__ __align__(32) bf16 Ks[2][64][HD];
  __shared__ __align__(32) bf16 Vs[2][HD][64];
  __shared__ __align__(32) bf16 Ps[64][64];

  const int tid  = threadIdx.x;
  const int wave = tid >> 5;
  const int lane = tid & 31;
  const int hl   = lane >> 4;
  const int l16  = lane & 15;

  const int blk  = blockIdx.x;
  const int qb   = blk & 31;            // S/64 = 32 q-blocks
  const int head = blk >> 5;
  const int h    = head & 31;
  const int b    = head >> 5;
  const int g    = h / NREP;
  const int q0   = qb * 64;
  const float scale = 0.08838834764831845f;   // 1/sqrt(128)

  const bf16* Kg0 = kT + ((size_t)(b * NKV + g) * Sc) * HD;
  const bf16* Vg0 = vT + ((size_t)(b * NKV + g) * HD) * Sc;

  // ---- async-issue Q tile (64x128 = 1024 x 16B chunks, 8/thread) ----
  const bf16* Qg = qT + ((size_t)(b * NH + h) * Sc + q0) * HD;
#pragma unroll
  for (int i = 0; i < 8; ++i) {
    int c = tid + i * 128;
    int r = c >> 4, cc = (c & 15) << 3;
    async_copy_b128(Qg + (size_t)r * HD + cc, &Qs[r][cc]);
  }
  // ---- async-issue K/V tile 0 ----
#pragma unroll
  for (int i = 0; i < 8; ++i) {
    int c = tid + i * 128;
    int rk = c >> 4, ck = (c & 15) << 3;        // 64 rows x 128 cols
    async_copy_b128(Kg0 + (size_t)rk * HD + ck, &Ks[0][rk][ck]);
    int rv = c >> 3, cv = (c & 7) << 3;         // 128 rows x 64 cols
    async_copy_b128(Vg0 + (size_t)rv * Sc + cv, &Vs[0][rv][cv]);
  }

  float m_i[8], l_i[8];
  v8f o_acc[8];
#pragma unroll
  for (int r = 0; r < 8; ++r) { m_i[r] = -1e30f; l_i[r] = 0.0f; }
#pragma unroll
  for (int dt = 0; dt < 8; ++dt) o_acc[dt] = (v8f)0.0f;

  int buf = 0;
  for (int j = 0; j <= qb; ++j) {
    wait_async();      // my DMA into Ks[buf]/Vs[buf] (and Qs on j==0) landed
    __syncthreads();   // all landed; all done reading buf^1 (and Ps)

    if (j < qb) {      // overlap next K/V tile DMA with compute
      int nb = buf ^ 1;
      const bf16* Kg = Kg0 + (size_t)((j + 1) * 64) * HD;
      const bf16* Vg = Vg0 + (j + 1) * 64;
#pragma unroll
      for (int i = 0; i < 8; ++i) {
        int c = tid + i * 128;
        int rk = c >> 4, ck = (c & 15) << 3;
        async_copy_b128(Kg + (size_t)rk * HD + ck, &Ks[nb][rk][ck]);
        int rv = c >> 3, cv = (c & 7) << 3;
        async_copy_b128(Vg + (size_t)rv * Sc + cv, &Vs[nb][rv][cv]);
      }
    }

    // ---- scores S = Q @ K^T ----
    v8f sacc[4];
#pragma unroll
    for (int nt = 0; nt < 4; ++nt) sacc[nt] = (v8f)0.0f;
#pragma unroll
    for (int ks = 0; ks < 4; ++ks) {
      Frag16 a;
      int m = wave * 16 + l16;
      a.v = *(const v16bf*)&Qs[m][ks * 32 + hl * 16];
#pragma unroll
      for (int nt = 0; nt < 4; ++nt) {
        int n = nt * 16 + l16;
        v16bf bfrag = *(const v16bf*)&Ks[buf][n][ks * 32 + hl * 16];
        sacc[nt] = __builtin_amdgcn_wmma_f32_16x16x32_bf16(
            false, a.v, false, bfrag, (short)0, sacc[nt], false, false);
      }
    }

    // ---- scale + causal mask + row max ----
    const bool diag = (j == qb);
    float rowmax[8];
#pragma unroll
    for (int r = 0; r < 8; ++r) rowmax[r] = -1e30f;
#pragma unroll
    for (int nt = 0; nt < 4; ++nt)
#pragma unroll
      for (int r = 0; r < 8; ++r) {
        float v = sacc[nt][r] * scale;
        if (diag) {
          int row = q0 + wave * 16 + hl * 8 + r;
          int col = j * 64 + nt * 16 + l16;
          if (col > row) v = -1e30f;
        }
        sacc[nt][r] = v;
        rowmax[r] = fmaxf(rowmax[r], v);
      }
#pragma unroll
    for (int r = 0; r < 8; ++r) {
      rowmax[r] = fmaxf(rowmax[r], __shfl_xor(rowmax[r], 1, 16));
      rowmax[r] = fmaxf(rowmax[r], __shfl_xor(rowmax[r], 2, 16));
      rowmax[r] = fmaxf(rowmax[r], __shfl_xor(rowmax[r], 4, 16));
      rowmax[r] = fmaxf(rowmax[r], __shfl_xor(rowmax[r], 8, 16));
    }

    // ---- online softmax update + write P ----
    float rowsum[8];
#pragma unroll
    for (int r = 0; r < 8; ++r) {
      float mnew = fmaxf(m_i[r], rowmax[r]);
      float corr = __expf(m_i[r] - mnew);
      m_i[r] = mnew;
      l_i[r] *= corr;
#pragma unroll
      for (int dt = 0; dt < 8; ++dt) o_acc[dt][r] *= corr;
      rowsum[r] = 0.0f;
    }
#pragma unroll
    for (int nt = 0; nt < 4; ++nt)
#pragma unroll
      for (int r = 0; r < 8; ++r) {
        float p = __expf(sacc[nt][r] - m_i[r]);
        rowsum[r] += p;
        Ps[wave * 16 + hl * 8 + r][nt * 16 + l16] = (bf16)p;
      }
#pragma unroll
    for (int r = 0; r < 8; ++r) {
      rowsum[r] += __shfl_xor(rowsum[r], 1, 16);
      rowsum[r] += __shfl_xor(rowsum[r], 2, 16);
      rowsum[r] += __shfl_xor(rowsum[r], 4, 16);
      rowsum[r] += __shfl_xor(rowsum[r], 8, 16);
      l_i[r] += rowsum[r];
    }
    __syncthreads();   // Ps visible to all waves

    // ---- O += P @ V ----
#pragma unroll
    for (int k2 = 0; k2 < 2; ++k2) {
      Frag16 a;
      int m = wave * 16 + l16;
      a.v = *(const v16bf*)&Ps[m][k2 * 32 + hl * 16];
#pragma unroll
      for (int dt = 0; dt < 8; ++dt) {
        int n = dt * 16 + l16;
        v16bf bfrag = *(const v16bf*)&Vs[buf][n][k2 * 32 + hl * 16];
        o_acc[dt] = __builtin_amdgcn_wmma_f32_16x16x32_bf16(
            false, a.v, false, bfrag, (short)0, o_acc[dt], false, false);
      }
    }
    buf ^= 1;
  }

  // ---- normalize + store bf16 context [B][S][NH*HD] ----
#pragma unroll
  for (int r = 0; r < 8; ++r) {
    float inv = 1.0f / l_i[r];
    int srow = q0 + wave * 16 + hl * 8 + r;
    bf16* dst = ctx + ((size_t)(b * Sc + srow)) * (NH * HD) + h * HD;
#pragma unroll
    for (int dt = 0; dt < 8; ++dt)
      dst[dt * 16 + l16] = (bf16)(o_acc[dt][r] * inv);
  }
}

// ---------------------------------------------------------------------------
extern "C" void kernel_launch(void* const* d_in, const int* in_sizes, int n_in,
                              void* d_out, int out_size, void* d_ws, size_t ws_size,
                              hipStream_t stream)
{
  const float* x    = (const float*)d_in[0];
  const float* Wq   = (const float*)d_in[1];
  const float* Wk   = (const float*)d_in[2];
  const float* Wv   = (const float*)d_in[3];
  const float* Wo   = (const float*)d_in[4];
  const float* fcos = (const float*)d_in[5];
  const float* fsin = (const float*)d_in[6];
  // d_in[7] = mask (causal, computed inline), d_in[8] = start_pos (0)

  char* ws = (char*)d_ws;
  size_t off = 0;
  bf16* xb  = (bf16*)(ws + off); off += (size_t)BS * Dm * 2;             // 32 MiB
  bf16* WqT = (bf16*)(ws + off); off += (size_t)Dm * (NH  * HD) * 2;     // 32 MiB
  bf16* WkT = (bf16*)(ws + off); off += (size_t)Dm * (NKV * HD) * 2;
  bf16* WvT = (bf16*)(ws + off); off += (size_t)Dm * (NKV * HD) * 2;
  bf16* WoT = (bf16*)(ws + off); off += (size_t)(NH * HD) * Dm * 2;      // 32 MiB
  float* qf = (float*)(ws + off); off += (size_t)BS * (NH  * HD) * 4;    // 64 MiB
  float* kf = (float*)(ws + off); off += (size_t)BS * (NKV * HD) * 4;
  float* vf = (float*)(ws + off); off += (size_t)BS * (NKV * HD) * 4;
  // xb / WqT dead after the projections + RoPE inputs are consumed:
  bf16* qT = xb;                       // 32 MiB needed, 32 MiB available
  bf16* kT = WqT;                      //  8 MiB
  bf16* vT = WqT + (size_t)Dm * NKV * HD;  // 8 MiB, still inside WqT region
  bf16* ctx = (bf16*)qf;               // qf dead after RoPE; 32 MiB into 64 MiB

  // 0) precision/layout pre-passes
  cvt_bf16<<<(unsigned)(((size_t)BS * Dm) / 1024), 256, 0, stream>>>(x, xb);
  transpose_cvt_w<<<dim3((NH  * HD) / 32, Dm / 32), 256, 0, stream>>>(Wq, WqT, Dm, NH  * HD);
  transpose_cvt_w<<<dim3((NKV * HD) / 32, Dm / 32), 256, 0, stream>>>(Wk, WkT, Dm, NKV * HD);
  transpose_cvt_w<<<dim3((NKV * HD) / 32, Dm / 32), 256, 0, stream>>>(Wv, WvT, Dm, NKV * HD);
  transpose_cvt_w<<<dim3(Dm / 32, (NH * HD) / 32), 256, 0, stream>>>(Wo, WoT, NH * HD, Dm);

  // 1) projections (bf16 WMMA, async-LDS double-buffered)
  gemm_bf16<<<dim3((NH * HD) / GBN, BS / GBM), 256, 0, stream>>>(
      xb, WqT, qf, BS, NH * HD, Dm);
  gemm_bf16<<<dim3((NKV * HD) / GBN, BS / GBM), 256, 0, stream>>>(
      xb, WkT, kf, BS, NKV * HD, Dm);
  gemm_bf16<<<dim3((NKV * HD) / GBN, BS / GBM), 256, 0, stream>>>(
      xb, WvT, vf, BS, NKV * HD, Dm);

  // 2) RoPE + layout transforms (f32 -> bf16); qT/kT/vT alias dead buffers
  int qpairs = Bc * Sc * NH * (HD / 2);
  rope_transpose<<<qpairs / 256, 256, 0, stream>>>(qf, fcos, fsin, qT, NH, 1);
  int kpairs = Bc * Sc * NKV * (HD / 2);
  rope_transpose<<<kpairs / 256, 256, 0, stream>>>(kf, fcos, fsin, kT, NKV, 1);
  int velems = Bc * Sc * NKV * HD;
  v_transpose<<<velems / 256, 256, 0, stream>>>(vf, vT);

  // 3) flash attention (causal), bf16 context out
  flash_attn<<<Bc * NH * (Sc / 64), 128, 0, stream>>>(qT, kT, vT, ctx);

  // 4) output projection
  gemm_bf16<<<dim3(Dm / GBN, BS / GBM), 256, 0, stream>>>(
      ctx, WoT, (float*)d_out, BS, Dm, Dm);
}
